// AttentionFlow_49143015801112
// MI455X (gfx1250) — matrix operand order repaired and checked by reading, not compiled, not run on hardware
//
#include <hip/hip_runtime.h>
#include <hip/hip_bf16.h>

// ---------------- constants from the reference ----------------
#define NN     50000          // NUM_NODES
#define DIN    256
#define D4     1024           // 4*D_IN
#define NQ     128            // N_QUERIES
#define PG     256            // PER_GROUP
#define E1CNT  (NQ*PG)        // 32768
#define E0CNT  32768
#define MAXE   64
#define NPRUNE (NQ*MAXE)      // 8192
#define RATIO  0.2f

// ---------------- types ----------------
typedef __attribute__((ext_vector_type(4)))  unsigned       v4u;
typedef __attribute__((ext_vector_type(16))) __bf16         v16bf;
typedef __attribute__((ext_vector_type(8)))  float          v8f;

union FragU { v4u q[2]; v16bf bf; };

// float -> bf16 bits (round to nearest even-ish)
__device__ __forceinline__ unsigned short f2bf(float f) {
    unsigned u = __float_as_uint(f);
    unsigned r = ((u >> 16) & 1u) + 0x7FFFu;
    return (unsigned short)((u + r) >> 16);
}
__device__ __forceinline__ float bf2f(unsigned short b) {
    return __uint_as_float(((unsigned)b) << 16);
}
// ordered-uint encoding for float atomicMax
__device__ __forceinline__ unsigned encf(float f) {
    unsigned u = __float_as_uint(f);
    return (u & 0x80000000u) ? ~u : (u | 0x80000000u);
}
__device__ __forceinline__ float decf(unsigned u) {
    unsigned b = (u & 0x80000000u) ? (u & 0x7FFFFFFFu) : ~u;
    return __uint_as_float(b);
}

// ---------------- transposing fp32 -> bf16 cast (1024x1024) ----------------
// dst[n][j] = bf16(src[j][n])
__global__ __launch_bounds__(256)
void transpose_cast_bf_kernel(const float* __restrict__ src,
                              unsigned short* __restrict__ dst) {
    __shared__ unsigned short tile[32][33];
    int bx = blockIdx.x * 32;          // src col block
    int by = blockIdx.y * 32;          // src row block
    int tx = threadIdx.x & 31;
    int ty = threadIdx.x >> 5;         // 0..7
    #pragma unroll
    for (int i = 0; i < 32; i += 8) {
        float v = src[(size_t)(by + ty + i) * D4 + bx + tx];
        tile[tx][ty + i] = f2bf(v);    // tile[col_local][row_local]
    }
    __syncthreads();
    #pragma unroll
    for (int i = 0; i < 32; i += 8) {
        dst[(size_t)(bx + ty + i) * D4 + by + tx] = tile[ty + i][tx];
    }
}

// ---------------- M = Wq^T @ Wk via WMMA (bf16 in/out) ----------------
// Ab: [1024 x 1024] row-major  (Wq^T as [n][j])
// Bb: [1024 x 1024] row-major  (Wk^T as [k2][j])  -> B[j][k2] fragment source
// Cb: [1024 x 1024] bf16 out   (M[n][k2])
__global__ __launch_bounds__(128)
void gemm_bf_wmma_kernel(const unsigned short* __restrict__ Ab,
                         const unsigned short* __restrict__ Bb,
                         unsigned short* __restrict__ Cb) {
    extern __shared__ unsigned short ldsbuf[];     // 4 * 16 * 1024 ushort = 128KB
    const int lane = threadIdx.x & 31;
    const int wave = threadIdx.x >> 5;
    const int rowBase = (blockIdx.x * 4 + wave) * 16;
    const int nb0 = blockIdx.y * 128;              // 8 column chunks of 128
    unsigned short* myL = ldsbuf + wave * 16 * D4;
    {
        const v4u* s4 = reinterpret_cast<const v4u*>(Ab + (size_t)rowBase * D4);
        v4u* d4 = reinterpret_cast<v4u*>(myL);
        #pragma unroll
        for (int i = 0; i < 64; ++i) d4[lane + 32 * i] = s4[lane + 32 * i];
    }
    __syncthreads();
    const int hf = lane >> 4, l15 = lane & 15, kbase = hf * 8;
    for (int nb = nb0; nb < nb0 + 128; nb += 16) {
        v8f acc = {0.f,0.f,0.f,0.f,0.f,0.f,0.f,0.f};
        const unsigned short* mrow = Bb + (size_t)(nb + l15) * D4 + hf * 8;
        for (int kb = 0; kb < D4; kb += 32) {
            FragU a, b;
            const unsigned short* ar = myL + l15 * D4 + kb + kbase;
            a.q[0] = *reinterpret_cast<const v4u*>(ar);
            a.q[1] = *reinterpret_cast<const v4u*>(ar + 16);
            const unsigned* mp = reinterpret_cast<const unsigned*>(mrow + kb);
            v4u b0 = { mp[0], mp[1], mp[2], mp[3] };
            v4u b1 = { mp[8], mp[9], mp[10], mp[11] };
            b.q[0] = b0; b.q[1] = b1;
            acc = __builtin_amdgcn_wmma_f32_16x16x32_bf16(false, a.bf, false, b.bf,
                                                          (short)0, acc, false, false);
        }
        int col = nb + l15;
        #pragma unroll
        for (int r = 0; r < 8; ++r) {
            int row = rowBase + r + 8 * hf;
            Cb[(size_t)row * D4 + col] = f2bf(acc[r]);
        }
    }
}

// ---------------- generic fp32 -> bf16 cast ----------------
__global__ void cast_bf_kernel(const float* __restrict__ src,
                               unsigned short* __restrict__ dst, size_t n) {
    size_t i = (size_t)blockIdx.x * blockDim.x + threadIdx.x;
    if (i < n) dst[i] = f2bf(src[i]);
}

// ---------------- gather edge features (bf16) ----------------
// Lfeat[e] = [repr[src], rel[e], qst[q], qr[q]]   Rfeat[e] = [repr[dst], rel[e], qst[q], qr[q]]
__global__ __launch_bounds__(256)
void gather_feat_kernel(const long long* __restrict__ edges,
                        const float* __restrict__ repr,
                        const float* __restrict__ rel,
                        const float* __restrict__ qst,
                        const float* __restrict__ qr,
                        unsigned short* __restrict__ Lf,
                        unsigned short* __restrict__ Rf) {
    int e = blockIdx.x, t = threadIdx.x;
    size_t eb = (size_t)e * 8;
    int q = (int)edges[eb + 0];
    int s = (int)edges[eb + 6];
    int d = (int)edges[eb + 7];
    size_t base = (size_t)e * D4;
    unsigned short lv = f2bf(repr[(size_t)s * DIN + t]);
    unsigned short rv = f2bf(repr[(size_t)d * DIN + t]);
    unsigned short re = f2bf(rel[(size_t)e * DIN + t]);
    unsigned short qs = f2bf(qst[(size_t)q * DIN + t]);
    unsigned short qv = f2bf(qr[(size_t)q * DIN + t]);
    Lf[base + t] = lv;  Lf[base + 256 + t] = re;  Lf[base + 512 + t] = qs;  Lf[base + 768 + t] = qv;
    Rf[base + t] = rv;  Rf[base + 256 + t] = re;  Rf[base + 512 + t] = qs;  Rf[base + 768 + t] = qv;
}

// ---------------- fused GEMM + row-dot:  logits = rowsum(Lf * (Rf @ M^T)) ----------------
// 4 waves / block, each wave owns one 16-edge tile; A tile staged in LDS (32KB/wave),
// compiler promotes it to the extended VGPR file; B (M matrix) streams from L2.
__global__ __launch_bounds__(128)
void logits_wmma_kernel(const unsigned short* __restrict__ Rf,
                        const unsigned short* __restrict__ Lf,
                        const unsigned short* __restrict__ Mb,
                        float* __restrict__ logits) {
    extern __shared__ unsigned short ldsbuf[];         // 64 rows * 1024 = 128KB
    const int lane = threadIdx.x & 31;
    const int wave = threadIdx.x >> 5;
    const int rowBase = (blockIdx.x * 4 + wave) * 16;
    unsigned short* myL = ldsbuf + wave * 16 * D4;

    // stage this wave's 16x1024 bf16 A tile into LDS (per-lane 64 x 16B)
    {
        const v4u* s4 = reinterpret_cast<const v4u*>(Rf + (size_t)rowBase * D4);
        v4u* d4 = reinterpret_cast<v4u*>(myL);
        #pragma unroll
        for (int i = 0; i < 64; ++i) d4[lane + 32 * i] = s4[lane + 32 * i];
    }
    __syncthreads();

    const int hf = lane >> 4;          // lane half (0/1)
    const int l15 = lane & 15;
    const int kbase = hf * 8;          // A-layout K offset for this half

    float psum[8];
    #pragma unroll
    for (int r = 0; r < 8; ++r) psum[r] = 0.f;

    for (int nb = 0; nb < D4; nb += 16) {
        v8f acc = {0.f,0.f,0.f,0.f,0.f,0.f,0.f,0.f};
        const unsigned short* mrow = Mb + (size_t)(nb + l15) * D4 + hf * 8;
        if (nb + 16 < D4)
            __builtin_prefetch(Mb + (size_t)(nb + 16 + l15) * D4 + hf * 16, 0, 1);
        for (int kb = 0; kb < D4; kb += 32) {
            FragU a, b;
            // A frag from LDS: lanes 0-15 K={kb..kb+7, kb+16..kb+23}; lanes 16-31 +8
            const unsigned short* ar = myL + l15 * D4 + kb + kbase;
            a.q[0] = *reinterpret_cast<const v4u*>(ar);
            a.q[1] = *reinterpret_cast<const v4u*>(ar + 16);
            // B frag: B[k][n] = M[n][k]; same K layout as A: {kb..kb+7, kb+16..kb+23}, +8 for upper lanes
            const unsigned* mp = reinterpret_cast<const unsigned*>(mrow + kb);
            v4u b0 = { mp[0], mp[1], mp[2], mp[3] };
            v4u b1 = { mp[8], mp[9], mp[10], mp[11] };
            b.q[0] = b0; b.q[1] = b1;
            acc = __builtin_amdgcn_wmma_f32_16x16x32_bf16(false, a.bf, false, b.bf,
                                                          (short)0, acc, false, false);
        }
        // epilogue: elementwise multiply by L tile, accumulate per-row partials
        #pragma unroll
        for (int r = 0; r < 8; ++r) {
            int row = rowBase + r + 8 * hf;
            float lv = bf2f(Lf[(size_t)row * D4 + nb + l15]);
            psum[r] += acc[r] * lv;
        }
    }
    // reduce the 16 column-lanes of each half
    #pragma unroll
    for (int r = 0; r < 8; ++r) {
        float p = psum[r];
        p += __shfl_xor(p, 1, 32);
        p += __shfl_xor(p, 2, 32);
        p += __shfl_xor(p, 4, 32);
        p += __shfl_xor(p, 8, 32);
        if (l15 == 0) logits[rowBase + r + 8 * hf] = p;
    }
}

// ---------------- segment softmax over src node ids (atomics) ----------------
__global__ void seg_max_kernel(const float* __restrict__ logits,
                               const long long* __restrict__ edges, int nE,
                               unsigned* __restrict__ segmax) {
    int i = blockIdx.x * blockDim.x + threadIdx.x;
    if (i >= nE) return;
    int s = (int)edges[(size_t)i * 8 + 6];
    atomicMax(&segmax[s], encf(logits[i]));
}
__global__ void seg_expsum_kernel(const float* __restrict__ logits,
                                  const long long* __restrict__ edges, int nE,
                                  const unsigned* __restrict__ segmax,
                                  float* __restrict__ soft, float* __restrict__ segsum) {
    int i = blockIdx.x * blockDim.x + threadIdx.x;
    if (i >= nE) return;
    int s = (int)edges[(size_t)i * 8 + 6];
    float e = __expf(logits[i] - decf(segmax[s]));
    soft[i] = e;
    atomicAdd(&segsum[s], e);
}
__global__ void seg_norm_kernel(const long long* __restrict__ edges, int nE,
                                const float* __restrict__ segsum,
                                float* __restrict__ soft,
                                const float* __restrict__ score,
                                float* __restrict__ target) {
    int i = blockIdx.x * blockDim.x + threadIdx.x;
    if (i >= nE) return;
    int s = (int)edges[(size_t)i * 8 + 6];
    float w = soft[i] / segsum[s];
    soft[i] = w;
    if (target) target[i] = w * score[s];
}

// ---------------- per-query top-64 of 256 (descending, iterative argmax) ----------------
__global__ __launch_bounds__(256)
void topk_kernel(const float* __restrict__ target, const float* __restrict__ soft,
                 const long long* __restrict__ edges1,
                 long long* __restrict__ out_pe, long long* __restrict__ out_oi,
                 float* __restrict__ psoft, float* __restrict__ ptarget) {
    __shared__ float val[256];
    __shared__ float rv[256];
    __shared__ int   ri[256];
    __shared__ int   win[MAXE];
    int q = blockIdx.x, t = threadIdx.x;
    val[t] = target[q * PG + t];
    __syncthreads();
    for (int it = 0; it < MAXE; ++it) {
        rv[t] = val[t]; ri[t] = t;
        __syncthreads();
        for (int s = 128; s > 0; s >>= 1) {
            if (t < s && rv[t + s] > rv[t]) { rv[t] = rv[t + s]; ri[t] = ri[t + s]; }
            __syncthreads();
        }
        if (t == 0) { win[it] = ri[0]; val[ri[0]] = -3.4e38f; }
        __syncthreads();
    }
    if (t < MAXE) {
        int oi = q * PG + win[t];
        int gi = q * MAXE + t;
        out_oi[gi]  = (long long)oi;
        psoft[gi]   = soft[oi];
        ptarget[gi] = target[oi];
        #pragma unroll
        for (int j = 0; j < 8; ++j)
            out_pe[(size_t)gi * 8 + j] = edges1[(size_t)oi * 8 + j];
    }
}

// ---------------- scatter kernels ----------------
__global__ void score_agg_kernel(const long long* __restrict__ pe, int nE,
                                 const float* __restrict__ ptarget,
                                 float* __restrict__ out_score) {
    int i = blockIdx.x * blockDim.x + threadIdx.x;
    if (i >= nE) return;
    atomicAdd(&out_score[(int)pe[(size_t)i * 8 + 7]], ptarget[i]);
}
__global__ void mark_src_kernel(const long long* __restrict__ edges, int nE,
                                int* __restrict__ isrc) {
    int i = blockIdx.x * blockDim.x + threadIdx.x;
    if (i >= nE) return;
    isrc[(int)edges[(size_t)i * 8 + 6]] = 1;
}
__global__ __launch_bounds__(256)
void repr_diag_kernel(const float* __restrict__ base, const int* __restrict__ isrc,
                      float* __restrict__ out) {
    int v = blockIdx.x, t = threadIdx.x;
    float dg = isrc[v] ? RATIO : 1.0f;
    out[(size_t)v * DIN + t] = dg * base[(size_t)v * DIN + t];
}
__global__ __launch_bounds__(256)
void repr_scatter_kernel(const long long* __restrict__ edges,
                         const float* __restrict__ w,
                         const float* __restrict__ base,
                         float* __restrict__ out) {
    int e = blockIdx.x, t = threadIdx.x;
    int s = (int)edges[(size_t)e * 8 + 6];
    int d = (int)edges[(size_t)e * 8 + 7];
    float c = (1.0f - RATIO) * w[e];
    atomicAdd(&out[(size_t)s * DIN + t], c * base[(size_t)d * DIN + t]);
}

// ---------------- final linear: leaky_relu(X @ Wl^T + bl) via WMMA ----------------
__global__ __launch_bounds__(128)
void linear_wmma_kernel(const unsigned short* __restrict__ Xb,
                        const unsigned short* __restrict__ Wb,   // Wl row-major [n][k]
                        const float* __restrict__ bias,
                        float* __restrict__ out, int M) {
    extern __shared__ unsigned short ldsbuf[];          // 4 * 16 * 256 ushort = 32KB
    const int lane = threadIdx.x & 31;
    const int wave = threadIdx.x >> 5;
    const int rowBase = (blockIdx.x * 4 + wave) * 16;
    unsigned short* myL = ldsbuf + wave * 16 * DIN;
    {
        v4u* d4 = reinterpret_cast<v4u*>(myL);
        #pragma unroll
        for (int i = 0; i < 8; ++i) {
            int cc = lane + 32 * i;                     // 256 chunks of 16B
            int rr = cc >> 4;
            int row = rowBase + rr; if (row > M - 1) row = M - 1;
            d4[cc] = reinterpret_cast<const v4u*>(Xb + (size_t)row * DIN)[cc & 15];
        }
    }
    __syncthreads();
    const int hf = lane >> 4, l15 = lane & 15, kbase = hf * 8;
    for (int nb = 0; nb < DIN; nb += 16) {
        v8f acc = {0.f,0.f,0.f,0.f,0.f,0.f,0.f,0.f};
        const unsigned short* mrow = Wb + (size_t)(nb + l15) * DIN + hf * 8;
        #pragma unroll
        for (int kb = 0; kb < DIN; kb += 32) {
            FragU a, b;
            const unsigned short* ar = myL + l15 * DIN + kb + kbase;
            a.q[0] = *reinterpret_cast<const v4u*>(ar);
            a.q[1] = *reinterpret_cast<const v4u*>(ar + 16);
            const unsigned* mp = reinterpret_cast<const unsigned*>(mrow + kb);
            v4u b0 = { mp[0], mp[1], mp[2], mp[3] };
            v4u b1 = { mp[8], mp[9], mp[10], mp[11] };
            b.q[0] = b0; b.q[1] = b1;
            acc = __builtin_amdgcn_wmma_f32_16x16x32_bf16(false, a.bf, false, b.bf,
                                                          (short)0, acc, false, false);
        }
        int col = nb + l15;
        float bv = bias[col];
        #pragma unroll
        for (int r = 0; r < 8; ++r) {
            int row = rowBase + r + 8 * hf;
            float v = acc[r] + bv;
            v = v > 0.f ? v : 0.01f * v;                // jax leaky_relu default slope
            if (row < M) out[(size_t)row * DIN + col] = v;
        }
    }
}

// ---------------- host-side orchestration ----------------
extern "C" void kernel_launch(void* const* d_in, const int* in_sizes, int n_in,
                              void* d_out, int out_size, void* d_ws, size_t ws_size,
                              hipStream_t stream) {
    (void)in_sizes; (void)n_in; (void)out_size; (void)ws_size;
    const float*     score_in = (const float*)d_in[0];
    const float*     repr_in  = (const float*)d_in[1];
    const long long* edges0   = (const long long*)d_in[2];
    const long long* edges1   = (const long long*)d_in[3];
    const float*     rel0     = (const float*)d_in[4];
    const float*     rel1     = (const float*)d_in[5];
    const float*     qst      = (const float*)d_in[6];
    const float*     qr       = (const float*)d_in[7];
    const float*     Wq       = (const float*)d_in[8];
    const float*     Wk       = (const float*)d_in[9];
    const float*     Wl       = (const float*)d_in[10];
    const float*     bl       = (const float*)d_in[11];

    char* ws = (char*)d_ws;
    size_t off = 0;
    auto wsalloc = [&](size_t bytes) -> char* {
        char* p = ws + off;
        off = (off + bytes + 255) & ~(size_t)255;
        return p;
    };
    unsigned short* WqTb    = (unsigned short*)wsalloc((size_t)D4 * D4 * 2);
    unsigned short* WkTb    = (unsigned short*)wsalloc((size_t)D4 * D4 * 2);
    unsigned short* Mb      = (unsigned short*)wsalloc((size_t)D4 * D4 * 2);
    unsigned short* Wlb     = (unsigned short*)wsalloc((size_t)DIN * DIN * 2);
    unsigned short* Lf      = (unsigned short*)wsalloc((size_t)E1CNT * D4 * 2);
    unsigned short* Rf      = (unsigned short*)wsalloc((size_t)E1CNT * D4 * 2);
    float*          logits  = (float*)wsalloc((size_t)E1CNT * 4);
    float*          soft    = (float*)wsalloc((size_t)E1CNT * 4);
    float*          target  = (float*)wsalloc((size_t)E1CNT * 4);
    unsigned*       segmax  = (unsigned*)wsalloc((size_t)NN * 4);
    float*          segsum  = (float*)wsalloc((size_t)NN * 4);
    int*            isrc    = (int*)wsalloc((size_t)NN * 4);
    float*          psoft   = (float*)wsalloc((size_t)NPRUNE * 4);
    float*          ptarget = (float*)wsalloc((size_t)NPRUNE * 4);
    float*          repr_a  = (float*)wsalloc((size_t)NN * DIN * 4);
    float*          repr_b  = (float*)wsalloc((size_t)NN * DIN * 4);
    unsigned short* Xb      = (unsigned short*)wsalloc((size_t)NN * DIN * 2);

    float*     out_score = (float*)d_out;
    float*     out_repr  = out_score + NN;
    long long* out_pe    = (long long*)(out_repr + (size_t)NN * DIN);
    long long* out_oi    = out_pe + (size_t)NPRUNE * 8;

    const size_t ldsBig    = (size_t)64 * D4 * 2;   // 128KB (logits / M-GEMM)
    const size_t ldsLinear = (size_t)64 * DIN * 2;  // 32KB

    // ---- M = Wq^T Wk via WMMA: transpose-cast both operands, then GEMM ----
    transpose_cast_bf_kernel<<<dim3(32, 32), 256, 0, stream>>>(Wq, WqTb);
    transpose_cast_bf_kernel<<<dim3(32, 32), 256, 0, stream>>>(Wk, WkTb);
    gemm_bf_wmma_kernel<<<dim3(16, 8), 128, ldsBig, stream>>>(WqTb, WkTb, Mb);
    cast_bf_kernel<<<(DIN * DIN + 255) / 256, 256, 0, stream>>>(Wl, Wlb, (size_t)DIN * DIN);

    // ---- edges1: scores, softmax, top-k, prune, score agg, repr update #1 ----
    gather_feat_kernel<<<E1CNT, 256, 0, stream>>>(edges1, repr_in, rel1, qst, qr, Lf, Rf);
    logits_wmma_kernel<<<E1CNT / 64, 128, ldsBig, stream>>>(Rf, Lf, Mb, logits);
    hipMemsetAsync(segmax, 0, (size_t)NN * 4, stream);
    hipMemsetAsync(segsum, 0, (size_t)NN * 4, stream);
    seg_max_kernel<<<(E1CNT + 255) / 256, 256, 0, stream>>>(logits, edges1, E1CNT, segmax);
    seg_expsum_kernel<<<(E1CNT + 255) / 256, 256, 0, stream>>>(logits, edges1, E1CNT, segmax, soft, segsum);
    seg_norm_kernel<<<(E1CNT + 255) / 256, 256, 0, stream>>>(edges1, E1CNT, segsum, soft, score_in, target);
    topk_kernel<<<NQ, 256, 0, stream>>>(target, soft, edges1, out_pe, out_oi, psoft, ptarget);
    hipMemsetAsync(out_score, 0, (size_t)NN * 4, stream);
    score_agg_kernel<<<(NPRUNE + 255) / 256, 256, 0, stream>>>(out_pe, NPRUNE, ptarget, out_score);
    hipMemsetAsync(isrc, 0, (size_t)NN * 4, stream);
    mark_src_kernel<<<(NPRUNE + 255) / 256, 256, 0, stream>>>(out_pe, NPRUNE, isrc);
    repr_diag_kernel<<<NN, 256, 0, stream>>>(repr_in, isrc, repr_a);
    repr_scatter_kernel<<<NPRUNE, 256, 0, stream>>>(out_pe, psoft, repr_in, repr_a);

    // ---- edges0: scores on updated repr, softmax, repr update #2 ----
    gather_feat_kernel<<<E0CNT, 256, 0, stream>>>(edges0, repr_a, rel0, qst, qr, Lf, Rf);
    logits_wmma_kernel<<<E0CNT / 64, 128, ldsBig, stream>>>(Rf, Lf, Mb, logits);
    hipMemsetAsync(segmax, 0, (size_t)NN * 4, stream);
    hipMemsetAsync(segsum, 0, (size_t)NN * 4, stream);
    seg_max_kernel<<<(E0CNT + 255) / 256, 256, 0, stream>>>(logits, edges0, E0CNT, segmax);
    seg_expsum_kernel<<<(E0CNT + 255) / 256, 256, 0, stream>>>(logits, edges0, E0CNT, segmax, soft, segsum);
    seg_norm_kernel<<<(E0CNT + 255) / 256, 256, 0, stream>>>(edges0, E0CNT, segsum, soft, nullptr, nullptr);
    hipMemsetAsync(isrc, 0, (size_t)NN * 4, stream);
    mark_src_kernel<<<(E0CNT + 255) / 256, 256, 0, stream>>>(edges0, E0CNT, isrc);
    repr_diag_kernel<<<NN, 256, 0, stream>>>(repr_a, isrc, repr_b);
    repr_scatter_kernel<<<E0CNT, 256, 0, stream>>>(edges0, soft, repr_a, repr_b);

    // ---- final linear + leaky relu ----
    cast_bf_kernel<<<((size_t)NN * DIN + 255) / 256, 256, 0, stream>>>(repr_b, Xb, (size_t)NN * DIN);
    linear_wmma_kernel<<<(NN + 63) / 64, 128, ldsLinear, stream>>>(Xb, Wlb, bl, out_repr, NN);
}
